// GraphDecoder_47717086658560
// MI455X (gfx1250) — matrix operand order repaired and checked by reading, not compiled, not run on hardware
//
#include <hip/hip_runtime.h>
#include <hip/hip_bf16.h>
#include <math.h>

#define B_  256
#define N_  2000
#define NP_ 2016          // padded node count for WMMA K-tiles (63*32)
#define E_  256
#define D_  768

typedef __attribute__((ext_vector_type(16))) __bf16 v16bf;
typedef __attribute__((ext_vector_type(8)))  float  v8f;

union BF16x16 { v16bf v; __bf16 h[16]; uint4 q[2]; };

// ---------------- Kernel 1: partial sums for graph mean -------------------
__global__ void k1_partial(const float* __restrict__ x, float* __restrict__ part) {
  int split = blockIdx.x, b = blockIdx.y, t = threadIdx.x;
  int n0 = split * (N_ / 8), n1 = n0 + (N_ / 8);
  const float* xb = x + (size_t)b * N_ * E_;
  float s = 0.f;
  for (int n = n0; n < n1; ++n) s += xb[(size_t)n * E_ + t];
  part[((size_t)b * 8 + split) * E_ + t] = s;
}

// ---------------- Kernel 2: per-batch context/q and folded score vectors --
__global__ void k2_prep(const float* __restrict__ part,
                        const float* __restrict__ first, const float* __restrict__ last,
                        const float* __restrict__ qw,  const float* __restrict__ in_b,
                        const float* __restrict__ kw,
                        const float* __restrict__ qwL, const float* __restrict__ in_bL,
                        const float* __restrict__ kwL,
                        float* __restrict__ Call) {
  __shared__ float ctx[3 * E_];
  __shared__ float qv[D_];
  __shared__ float qLv[E_];
  int b = blockIdx.x, t = threadIdx.x;
  float ge = 0.f;
  for (int s = 0; s < 8; ++s) ge += part[((size_t)b * 8 + s) * E_ + t];
  ctx[t]          = ge * (1.0f / N_);
  ctx[E_ + t]     = first[t];
  ctx[2 * E_ + t] = last[t];
  __syncthreads();
  for (int r = 0; r < 3; ++r) {                       // q = Wq*context + bq
    int j = t + r * E_;
    float acc = in_b[j];
    const float* wr = qw + (size_t)j * D_;
    for (int d = 0; d < D_; ++d) acc += wr[d] * ctx[d];
    qv[j] = acc;
  }
  {                                                   // qL = WqL*last + bqL
    float acc = in_bL[t];
    const float* wr = qwL + (size_t)t * E_;
    for (int d = 0; d < E_; ++d) acc += wr[d] * ctx[2 * E_ + d];
    qLv[t] = acc;
  }
  __syncthreads();
  const float sc_m = 0.10206207261596575f;            // 1/sqrt(96)
  const float sc_l = 0.17677669529663687f;            // 1/sqrt(32)
  for (int m = 0; m < 8; ++m) {                       // C[m,e] = sum_d q_h[d]*Wk[h*96+d,e]
    float acc = 0.f;
    for (int d = 0; d < 96; ++d) acc += qv[m * 96 + d] * kw[(size_t)(m * 96 + d) * E_ + t];
    Call[((size_t)b * 16 + m) * E_ + t] = acc * sc_m;
  }
  for (int h = 0; h < 8; ++h) {
    float acc = 0.f;
    for (int d = 0; d < 32; ++d) acc += qLv[h * 32 + d] * kwL[(size_t)(h * 32 + d) * E_ + t];
    Call[((size_t)b * 16 + 8 + h) * E_ + t] = acc * sc_l;
  }
}

// ---------------- Kernel 3: scores[b,16,N] = C_b (16x256) * X_b^T via WMMA
__global__ void __launch_bounds__(32)
k3_scores(const float* __restrict__ x, const float* __restrict__ Call,
          float* __restrict__ scores) {
  int tile = blockIdx.x, b = blockIdx.y, lane = threadIdx.x;
  int n0 = tile * 16;
  int mrow = lane & 15, hi = lane >> 4;
  const float* crow = Call + ((size_t)b * 16 + mrow) * E_;
  const float* xrow = x + ((size_t)b * N_ + (n0 + (lane & 15))) * E_;
  v8f acc{};
  for (int k0 = 0; k0 < E_; k0 += 32) {
    BF16x16 A, Bf;
    int kb = k0 + (hi ? 8 : 0);
    float4 a0 = *(const float4*)(crow + kb);
    float4 a1 = *(const float4*)(crow + kb + 4);
    float4 a2 = *(const float4*)(crow + kb + 16);
    float4 a3 = *(const float4*)(crow + kb + 20);
    A.h[0]=(__bf16)a0.x; A.h[1]=(__bf16)a0.y; A.h[2]=(__bf16)a0.z; A.h[3]=(__bf16)a0.w;
    A.h[4]=(__bf16)a1.x; A.h[5]=(__bf16)a1.y; A.h[6]=(__bf16)a1.z; A.h[7]=(__bf16)a1.w;
    A.h[8]=(__bf16)a2.x; A.h[9]=(__bf16)a2.y; A.h[10]=(__bf16)a2.z; A.h[11]=(__bf16)a2.w;
    A.h[12]=(__bf16)a3.x;A.h[13]=(__bf16)a3.y;A.h[14]=(__bf16)a3.z; A.h[15]=(__bf16)a3.w;
    int fb = k0 + (hi << 3);
    const float4* xr = (const float4*)(xrow + fb);
#pragma unroll
    for (int i = 0; i < 4; ++i) {
      float4 tv = xr[i + (i & 2)];
      Bf.h[4*i+0]=(__bf16)tv.x; Bf.h[4*i+1]=(__bf16)tv.y;
      Bf.h[4*i+2]=(__bf16)tv.z; Bf.h[4*i+3]=(__bf16)tv.w;
    }
    acc = __builtin_amdgcn_wmma_f32_16x16x32_bf16(false, A.v, false, Bf.v,
                                                  (short)0, acc, false, false);
  }
  int ncol = n0 + (lane & 15);
#pragma unroll
  for (int r = 0; r < 8; ++r) {
    int m = r + (hi ? 8 : 0);
    scores[((size_t)b * 16 + m) * N_ + ncol] = acc[r];
  }
}

// ---------------- Kernel 4: masked softmax -> bf16 weights (padded to 2016)
__global__ void k4_softmax(const float* __restrict__ scores,
                           const unsigned char* __restrict__ mask,
                           __bf16* __restrict__ wbf) {
  __shared__ float red[256];
  int row = blockIdx.x, b = blockIdx.y, t = threadIdx.x;
  const float* srow = scores + ((size_t)b * 16 + row) * N_;
  const unsigned char* mrow = mask + (size_t)b * N_;
  float mx = -INFINITY;
  for (int n = t; n < N_; n += 256) if (!mrow[n]) mx = fmaxf(mx, srow[n]);
  red[t] = mx; __syncthreads();
  for (int s = 128; s; s >>= 1) { if (t < s) red[t] = fmaxf(red[t], red[t + s]); __syncthreads(); }
  mx = red[0]; __syncthreads();
  float sm = 0.f;
  for (int n = t; n < N_; n += 256) if (!mrow[n]) sm += expf(srow[n] - mx);
  red[t] = sm; __syncthreads();
  for (int s = 128; s; s >>= 1) { if (t < s) red[t] += red[t + s]; __syncthreads(); }
  float inv = 1.0f / red[0];
  __bf16* wrow = wbf + ((size_t)b * 16 + row) * NP_;
  for (int n = t; n < NP_; n += 256) {
    float w = 0.f;
    if (n < N_ && !mrow[n]) w = expf(srow[n] - mx) * inv;
    wrow[n] = (__bf16)w;
  }
}

// ---------------- Kernel 5: s[b,16,256] = W_b (16xN) * X_b via WMMA -------
__global__ void __launch_bounds__(512)
k5_wsum(const float* __restrict__ x, const __bf16* __restrict__ wbf,
        float* __restrict__ sbuf) {
  int b = blockIdx.x;
  int wave = threadIdx.x >> 5, lane = threadIdx.x & 31;
  int f0 = wave * 16, hi = lane >> 4, mrow = lane & 15;
  const __bf16* wrow = wbf + ((size_t)b * 16 + mrow) * NP_;
  int feat = f0 + (lane & 15);
  const float* xb = x + (size_t)b * N_ * E_;
  v8f acc{};
  for (int k0 = 0; k0 < NP_; k0 += 32) {
    BF16x16 A, Bf;
    int kb = k0 + (hi ? 8 : 0);
    A.q[0] = *(const uint4*)(wrow + kb);
    A.q[1] = *(const uint4*)(wrow + kb + 16);
    int nb = k0 + (hi << 3);
#pragma unroll
    for (int i = 0; i < 16; ++i) {
      int node = nb + i + (i & 8); if (node >= N_) node = N_ - 1;   // weight is 0 in pad
      Bf.h[i] = (__bf16)xb[(size_t)node * E_ + feat];
    }
    acc = __builtin_amdgcn_wmma_f32_16x16x32_bf16(false, A.v, false, Bf.v,
                                                  (short)0, acc, false, false);
  }
#pragma unroll
  for (int r = 0; r < 8; ++r) {
    int m = r + (hi ? 8 : 0);
    sbuf[((size_t)b * 16 + m) * E_ + feat] = acc[r];
  }
}

// ---------------- Kernel 6: per-batch projections, load%, g ---------------
__global__ void k6_finish(const float* __restrict__ sbuf,
                          const float* __restrict__ vw,  const float* __restrict__ in_b,
                          const float* __restrict__ ow,  const float* __restrict__ ob,
                          const float* __restrict__ aow, const float* __restrict__ kp,
                          const float* __restrict__ vwL, const float* __restrict__ in_bL,
                          const float* __restrict__ owL, const float* __restrict__ obL,
                          const float* __restrict__ loadw,
                          float* __restrict__ g, float* __restrict__ out) {
  __shared__ float o[D_], ao[D_], qatt[E_], oL[E_], aoL[E_], red[256];
  int b = blockIdx.x, t = threadIdx.x;
  const float* sb = sbuf + (size_t)b * 16 * E_;
  for (int r = 0; r < 3; ++r) {                 // o = Wv-heads * s_main + bv
    int j = t + r * E_, h = j / 96;
    float acc = in_b[2 * D_ + j];
    const float* wr = vw + (size_t)j * E_;
    const float* sh = sb + h * E_;
    for (int e = 0; e < E_; ++e) acc += wr[e] * sh[e];
    o[j] = acc;
  }
  {                                             // oL = WvL-heads * s_load + bvL
    int h = t / 32;
    float acc = in_bL[2 * E_ + t];
    const float* wr = vwL + (size_t)t * E_;
    const float* sh = sb + (8 + h) * E_;
    for (int e = 0; e < E_; ++e) acc += wr[e] * sh[e];
    oL[t] = acc;
  }
  __syncthreads();
  for (int r = 0; r < 3; ++r) {                 // ao = Wout*o + bout
    int j = t + r * E_;
    float acc = ob[j];
    const float* wr = ow + (size_t)j * D_;
    for (int d = 0; d < D_; ++d) acc += wr[d] * o[d];
    ao[j] = acc;
  }
  {                                             // aoL = WoutL*oL + boutL
    float acc = obL[t];
    const float* wr = owL + (size_t)t * E_;
    for (int d = 0; d < E_; ++d) acc += wr[d] * oL[d];
    aoL[t] = acc;
  }
  __syncthreads();
  {                                             // q_att = att_out_w * ao
    float acc = 0.f;
    const float* wr = aow + (size_t)t * D_;
    for (int d = 0; d < D_; ++d) acc += wr[d] * ao[d];
    qatt[t] = acc;
  }
  red[t] = loadw[t] * aoL[t];                   // l = load_out_w . aoL
  __syncthreads();
  for (int s2 = 128; s2; s2 >>= 1) { if (t < s2) red[t] += red[t + s2]; __syncthreads(); }
  if (t == 0) out[B_ + b] = 1.0f / (1.0f + expf(-red[0]));   // load_percent
  float acc = 0.f;                               // g = kp^T q_att, /16 folded
  for (int j = 0; j < E_; ++j) acc += qatt[j] * kp[(size_t)j * E_ + t];
  g[(size_t)b * E_ + t] = acc * 0.0625f;
}

// ---------------- Kernel 7: u[b,n] = tanh(g.x)*C, masked ------------------
__global__ void k7_u(const float* __restrict__ x, const float* __restrict__ g,
                     const unsigned char* __restrict__ mask, const int* __restrict__ Cp,
                     float* __restrict__ u) {
  __shared__ float gs[E_];
  int tile = blockIdx.x, b = blockIdx.y, t = threadIdx.x;
  gs[t] = g[(size_t)b * E_ + t];
  __syncthreads();
  float Cv = (float)Cp[0];
  int node = tile * 16 + (t >> 4);
  int fs = (t & 15) * 16;
  const float* xr = x + ((size_t)b * N_ + node) * E_ + fs;
  float p = 0.f;
#pragma unroll
  for (int i = 0; i < 16; ++i) p += gs[fs + i] * xr[i];
  p += __shfl_xor(p, 1); p += __shfl_xor(p, 2);
  p += __shfl_xor(p, 4); p += __shfl_xor(p, 8);
  if ((t & 15) == 0) {
    float val = mask[(size_t)b * N_ + node] ? -INFINITY : tanhf(p) * Cv;
    u[(size_t)b * N_ + node] = val;
  }
}

// ---------------- Kernel 8: threefry gumbel sample + log_softmax gather ---
__device__ __forceinline__ void threefry2x32(unsigned c0, unsigned c1,
                                             unsigned& o0, unsigned& o1) {
  const unsigned k0 = 0u, k1 = 42u;              // jax.random.key(42)
  const unsigned ks2 = k0 ^ k1 ^ 0x1BD11BDAu;
  unsigned x0 = c0 + k0, x1 = c1 + k1;
#define TFR(r) { x0 += x1; x1 = (x1 << (r)) | (x1 >> (32 - (r))); x1 ^= x0; }
  TFR(13) TFR(15) TFR(26) TFR(6)   x0 += k1;  x1 += ks2 + 1u;
  TFR(17) TFR(29) TFR(16) TFR(24)  x0 += ks2; x1 += k0 + 2u;
  TFR(13) TFR(15) TFR(26) TFR(6)   x0 += k0;  x1 += k1 + 3u;
  TFR(17) TFR(29) TFR(16) TFR(24)  x0 += k1;  x1 += ks2 + 4u;
  TFR(13) TFR(15) TFR(26) TFR(6)   x0 += ks2; x1 += k0 + 5u;
#undef TFR
  o0 = x0; o1 = x1;
}

__global__ void k8_sample(const float* __restrict__ u, float* __restrict__ out) {
  __shared__ float rz[256]; __shared__ int ri[256]; __shared__ float rm[256];
  int b = blockIdx.x, t = threadIdx.x;
  const float* ur = u + (size_t)b * N_;
  const unsigned HALF = (unsigned)(B_ * N_) / 2u;
  float bestz = -INFINITY; int besti = 0; float mx = -INFINITY;
  for (int n = t; n < N_; n += 256) {
    float uv = ur[n];
    unsigned e = (unsigned)(b * N_ + n);
    unsigned p = (e < HALF) ? e : (e - HALF);
    unsigned o0, o1; threefry2x32(p, p + HALF, o0, o1);
    unsigned bits = (e < HALF) ? o0 : o1;
    float f = __uint_as_float((bits >> 9) | 0x3f800000u) - 1.0f;
    const float TINY = 1.17549435e-38f;
    f = f * (1.0f - TINY) + TINY; if (f < TINY) f = TINY;
    float z = uv + (-logf(-logf(f)));
    if (z > bestz) { bestz = z; besti = n; }
    mx = fmaxf(mx, uv);
  }
  rz[t] = bestz; ri[t] = besti; rm[t] = mx; __syncthreads();
  for (int s = 128; s; s >>= 1) {
    if (t < s) {
      if (rz[t + s] > rz[t] || (rz[t + s] == rz[t] && ri[t + s] < ri[t])) {
        rz[t] = rz[t + s]; ri[t] = ri[t + s];
      }
      rm[t] = fmaxf(rm[t], rm[t + s]);
    }
    __syncthreads();
  }
  mx = rm[0]; int idx = ri[0]; __syncthreads();
  float sm = 0.f;
  for (int n = t; n < N_; n += 256) sm += expf(ur[n] - mx);
  rm[t] = sm; __syncthreads();
  for (int s = 128; s; s >>= 1) { if (t < s) rm[t] += rm[t + s]; __syncthreads(); }
  if (t == 0) {
    float lse = mx + logf(rm[0]);
    out[b] = (float)idx;                // nn_idx
    out[2 * B_ + b] = ur[idx] - lse;    // log_prob
  }
}

extern "C" void kernel_launch(void* const* d_in, const int* in_sizes, int n_in,
                              void* d_out, int out_size, void* d_ws, size_t ws_size,
                              hipStream_t stream) {
  (void)in_sizes; (void)n_in; (void)out_size; (void)ws_size;
  const float*         x          = (const float*)d_in[0];
  const unsigned char* mask       = (const unsigned char*)d_in[1];
  const int*           Cp         = (const int*)d_in[2];
  const float*         first      = (const float*)d_in[3];
  const float*         last       = (const float*)d_in[4];
  const float*         kp_w       = (const float*)d_in[5];
  const float*         att_out_w  = (const float*)d_in[6];
  const float*         load_out_w = (const float*)d_in[7];
  const float*         attn_q_w   = (const float*)d_in[8];
  const float*         attn_k_w   = (const float*)d_in[9];
  const float*         attn_v_w   = (const float*)d_in[10];
  const float*         attn_in_b  = (const float*)d_in[11];
  const float*         attn_out_w = (const float*)d_in[12];
  const float*         attn_out_b = (const float*)d_in[13];
  const float*         attnL_q_w  = (const float*)d_in[14];
  const float*         attnL_k_w  = (const float*)d_in[15];
  const float*         attnL_v_w  = (const float*)d_in[16];
  const float*         attnL_in_b = (const float*)d_in[17];
  const float*         attnL_out_w= (const float*)d_in[18];
  const float*         attnL_out_b= (const float*)d_in[19];

  char* w = (char*)d_ws;
  const size_t SZ_PART = (size_t)B_ * 8 * E_ * 4;     // 2 MB
  const size_t SZ_C    = (size_t)B_ * 16 * E_ * 4;    // 4 MB
  const size_t SZ_SC   = (size_t)B_ * 16 * N_ * 4;    // 32 MB
  const size_t SZ_W    = (size_t)B_ * 16 * NP_ * 2;   // 16.5 MB
  const size_t SZ_S    = (size_t)B_ * 16 * E_ * 4;    // 4 MB
  const size_t SZ_G    = (size_t)B_ * E_ * 4;         // 0.25 MB
  float*          part   = (float*)(w);
  float*          Call   = (float*)(w + SZ_PART);
  float*          scores = (float*)(w + SZ_PART + SZ_C);
  __bf16*         wbf    = (__bf16*)(w + SZ_PART + SZ_C + SZ_SC);
  float*          sbuf   = (float*)(w + SZ_PART + SZ_C + SZ_SC + SZ_W);
  float*          gbuf   = (float*)(w + SZ_PART + SZ_C + SZ_SC + SZ_W + SZ_S);
  float*          ubuf   = (float*)(w + SZ_PART + SZ_C + SZ_SC + SZ_W + SZ_S + SZ_G);
  float*          outf   = (float*)d_out;

  k1_partial<<<dim3(8, B_), 256, 0, stream>>>(x, part);
  k2_prep<<<B_, 256, 0, stream>>>(part, first, last, attn_q_w, attn_in_b, attn_k_w,
                                  attnL_q_w, attnL_in_b, attnL_k_w, Call);
  k3_scores<<<dim3(N_ / 16, B_), 32, 0, stream>>>(x, Call, scores);
  k4_softmax<<<dim3(16, B_), 256, 0, stream>>>(scores, mask, wbf);
  k5_wsum<<<B_, 512, 0, stream>>>(x, wbf, sbuf);
  k6_finish<<<B_, 256, 0, stream>>>(sbuf, attn_v_w, attn_in_b, attn_out_w, attn_out_b,
                                    att_out_w, kp_w, attnL_v_w, attnL_in_b,
                                    attnL_out_w, attnL_out_b, load_out_w, gbuf, outf);
  k7_u<<<dim3(N_ / 16, B_), 256, 0, stream>>>(x, gbuf, mask, Cp, ubuf);
  k8_sample<<<B_, 256, 0, stream>>>(ubuf, outf);
}
